// GlobalCapsulePooling_32392643346834
// MI455X (gfx1250) — matrix operand order repaired and checked by reading, not compiled, not run on hardware
//
#include <hip/hip_runtime.h>
#include <math.h>

typedef __attribute__((ext_vector_type(16))) __bf16 v16bf;
typedef __attribute__((ext_vector_type(8)))  float  v8f;

#define NN 100000
#define NE 1600000
#define NG 256
#define TC 16
#define CH 64
#define COLS 1024   // TC*CH
#define BN_EPSF 1e-5f

// ---------- helpers ----------
__device__ inline unsigned short f2bf(float f) {
  unsigned int u = __float_as_uint(f);
  unsigned int r = (u + 0x7FFFu + ((u >> 16) & 1u)) >> 16;
  return (unsigned short)r;
}

// Compute one 16x16 U-tile (u = h_tile @ W + bias) with two bf16 WMMAs (K=64).
// hT: LDS tile 16 nodes x 64 ch (f32).
// Wv: W pre-swizzled into per-lane WMMA B fragments: [ct][ko][lane] -> v16bf.
__device__ inline v8f wmma_tile(const float* hT, const v16bf* __restrict__ Wv,
                                const float* __restrict__ bias, int ct, int lane) {
  const int half = lane >> 4;
  const int mrow = lane & 15;
  const int col = ct * 16 + mrow;
  v16bf a0, a1;
#pragma unroll
  for (int j = 0; j < 16; ++j) {
    int i = j >> 1, sub = j & 1;
    // A (16x32, 16-bit) layout: K = half*16 + j (matches B swizzle)
    int ka = half * 16 + (i << 1) + sub;
    a0[j] = (__bf16)hT[mrow * 64 + ka];
    a1[j] = (__bf16)hT[mrow * 64 + 32 + ka];
  }
  v16bf b0 = Wv[(ct * 2 + 0) * 32 + lane];   // one 32B vector load per fragment
  v16bf b1 = Wv[(ct * 2 + 1) * 32 + lane];
  float bv = bias[col];
  v8f acc = {bv, bv, bv, bv, bv, bv, bv, bv};
  acc = __builtin_amdgcn_wmma_f32_16x16x32_bf16(false, a0, false, b0, (short)0, acc, false, false);
  acc = __builtin_amdgcn_wmma_f32_16x16x32_bf16(false, a1, false, b1, (short)0, acc, false, false);
  return acc;
}

// ---------- BN ----------
__global__ __launch_bounds__(256) void k_bn_stats(const float* __restrict__ x,
                                                  float* __restrict__ partial) {
  int tid = threadIdx.x;
  int c = tid & 63;
  float s1 = 0.f, s2 = 0.f;
  for (int r = blockIdx.x * 4 + (tid >> 6); r < NN; r += gridDim.x * 4) {
    float v = x[r * CH + c];
    s1 += v; s2 += v * v;
  }
  atomicAdd(&partial[c], s1);
  atomicAdd(&partial[64 + c], s2);
}

__global__ void k_bn_final(const float* __restrict__ partial,
                           const float* __restrict__ gamma, const float* __restrict__ beta,
                           float* __restrict__ scale, float* __restrict__ shiftv) {
  int c = threadIdx.x;
  if (c < CH) {
    float mu = partial[c] / (float)NN;
    float var = partial[64 + c] / (float)NN - mu * mu;
    float sc = gamma[c] * rsqrtf(var + BN_EPSF);
    scale[c] = sc;
    shiftv[c] = beta[c] - mu * sc;
  }
}

__global__ __launch_bounds__(256) void k_xn(const float* __restrict__ x,
                                            const float* __restrict__ scale,
                                            const float* __restrict__ shiftv,
                                            float* __restrict__ xn) {
  int id = blockIdx.x * 256 + threadIdx.x;   // NN*CH total, exact multiple
  int c = id & 63;
  xn[id] = x[id] * scale[c] + shiftv[c];
}

// ---------- GCN normalization + aggregation ----------
__global__ __launch_bounds__(256) void k_deg_init(float* __restrict__ deg) {
  int n = blockIdx.x * 256 + threadIdx.x;
  if (n < NN) deg[n] = 1.0f;   // self-loop weight
}

__global__ __launch_bounds__(256) void k_edge_deg(const int* __restrict__ row,
                                                  const int* __restrict__ col,
                                                  const float* __restrict__ w,
                                                  float* __restrict__ deg,
                                                  float* __restrict__ ndeg) {
  int e = blockIdx.x * 256 + threadIdx.x;    // NE exact multiple
  atomicAdd(&deg[col[e]], w[e]);
  atomicAdd(&ndeg[row[e]], w[e]);            // b_log init = out-degree over rows
}

__global__ __launch_bounds__(256) void k_dinv(const float* __restrict__ deg,
                                              float* __restrict__ dinv) {
  int n = blockIdx.x * 256 + threadIdx.x;
  if (n < NN) dinv[n] = deg[n] > 0.f ? rsqrtf(deg[n]) : 0.f;
}

__global__ __launch_bounds__(256) void k_hself(const float* __restrict__ xn,
                                               const float* __restrict__ dinv,
                                               float* __restrict__ h) {
  int id = blockIdx.x * 256 + threadIdx.x;
  int n = id >> 6;
  float dd = dinv[n];
  h[id] = dd * dd * xn[id];
}

__global__ __launch_bounds__(256) void k_hscatter(const int* __restrict__ row,
                                                  const int* __restrict__ col,
                                                  const float* __restrict__ w,
                                                  const float* __restrict__ dinv,
                                                  const float* __restrict__ xn,
                                                  float* __restrict__ h) {
  long id = (long)blockIdx.x * 256 + threadIdx.x;  // NE*CH total
  int e = (int)(id >> 6), c = (int)(id & 63);
  int r = row[e], cl = col[e];
  float nw = dinv[r] * w[e] * dinv[cl];
  atomicAdd(&h[cl * CH + c], nw * xn[r * CH + c]);
}

// ---------- weight conversion into WMMA-B-swizzled bf16 ----------
// Target: Wswz[((ct*2 + ko)*32 + lane)*16 + j] = bf16(Wflat[kb][col])
//   half=lane>>4, mrow=lane&15, i=j>>1, sub=j&1
//   kb = ko*32 + half*16 + 2i + sub ; col = ct*16 + mrow
//   Wflat[k][t*64+o] = W[t][k][o]
__global__ __launch_bounds__(256) void k_wconv(const float* __restrict__ W,
                                               unsigned short* __restrict__ Wswz) {
  int id = blockIdx.x * 256 + threadIdx.x;   // 64*2*32*16 = 65536
  int j    = id & 15;
  int lane = (id >> 4) & 31;
  int ko   = (id >> 9) & 1;
  int ct   = id >> 10;
  int half = lane >> 4, mrow = lane & 15;
  int i = j >> 1, sub = j & 1;
  int kb = ko * 32 + half * 16 + (i << 1) + sub;
  int colg = ct * 16 + mrow;
  int t = colg >> 6, o = colg & 63;
  Wswz[id] = f2bf(W[t * (CH * CH) + kb * CH + o]);
}

__global__ __launch_bounds__(256) void k_binit(const float* __restrict__ ndeg,
                                               float* __restrict__ blog) {
  int id = blockIdx.x * 256 + threadIdx.x;   // NN*TC
  blog[id] = ndeg[id >> 4];
}

__global__ __launch_bounds__(256) void k_xsum(const float* __restrict__ xn,
                                              const int* __restrict__ batch,
                                              float* __restrict__ xsum,
                                              float* __restrict__ cnt) {
  int id = blockIdx.x * 256 + threadIdx.x;   // NN*CH
  int n = id >> 6, c = id & 63;
  int g = batch[n];
  atomicAdd(&xsum[g * CH + c], xn[id]);
  if (c == 0) atomicAdd(&cnt[g], 1.0f);
}

// ---------- routing: s = segsum(softmax(b) * U) with U recomputed via WMMA ----------
__global__ __launch_bounds__(256) void k_route_s(const float* __restrict__ h,
                                                 const v16bf* __restrict__ Wv,
                                                 const float* __restrict__ bias,
                                                 const float* __restrict__ blog,
                                                 const int* __restrict__ batch,
                                                 float* __restrict__ s) {
  __shared__ float hT[16 * 64];
  __shared__ float cT[16 * 16];
  __shared__ int gT[16];
  __shared__ int uni;
  const int tid = threadIdx.x;
  const int nodebase = blockIdx.x * 16;
  for (int i = tid; i < 16 * 64; i += 256) hT[i] = h[nodebase * 64 + i];
  cT[tid] = blog[nodebase * 16 + tid];
  if (tid < 16) gT[tid] = batch[nodebase + tid];
  __syncthreads();
  if (tid < 16) {                       // per-node softmax over 16 capsules
    float m = -1e30f;
    for (int k = 0; k < 16; ++k) m = fmaxf(m, cT[tid * 16 + k]);
    float sum = 0.f;
    for (int k = 0; k < 16; ++k) sum += __expf(cT[tid * 16 + k] - m);
    float inv = 1.f / sum;
    for (int k = 0; k < 16; ++k) cT[tid * 16 + k] = __expf(cT[tid * 16 + k] - m) * inv;
  }
  if (tid == 0) {
    int u = 1;
    for (int i = 1; i < 16; ++i) u &= (gT[i] == gT[0]);
    uni = u;
  }
  __syncthreads();
  const int lane = tid & 31;
  const int wave = tid >> 5;
  const int half = lane >> 4;
  const int mrow = lane & 15;
  for (int q = 0; q < 8; ++q) {
    int ct = wave * 8 + q;              // 8 waves x 8 col-tiles = 64 tiles (1024 cols)
    int col = ct * 16 + mrow;
    int cap = ct >> 2;                  // 64 cols per capsule; tile never straddles
    v8f acc = wmma_tile(hT, Wv, bias, ct, lane);
    if (uni) {
      float p = 0.f;
#pragma unroll
      for (int r = 0; r < 8; ++r) p += acc[r] * cT[(r + half * 8) * 16 + cap];
      p += __shfl_xor(p, 16, 32);       // merge M=0..7 (lanes<16) with M=8..15
      if (half == 0) atomicAdd(&s[gT[0] * COLS + col], p);
    } else {
#pragma unroll
      for (int r = 0; r < 8; ++r) {
        int M = r + half * 8;
        atomicAdd(&s[gT[M] * COLS + col], acc[r] * cT[M * 16 + cap]);
      }
    }
  }
}

// ---------- routing: b += sum_d U * v[batch] ----------
__global__ __launch_bounds__(256) void k_route_b(const float* __restrict__ h,
                                                 const v16bf* __restrict__ Wv,
                                                 const float* __restrict__ bias,
                                                 const float* __restrict__ v,
                                                 const int* __restrict__ batch,
                                                 float* __restrict__ blog) {
  __shared__ float hT[16 * 64];
  __shared__ int gT[16];
  const int tid = threadIdx.x;
  const int nodebase = blockIdx.x * 16;
  for (int i = tid; i < 16 * 64; i += 256) hT[i] = h[nodebase * 64 + i];
  if (tid < 16) gT[tid] = batch[nodebase + tid];
  __syncthreads();
  const int lane = tid & 31;
  const int wave = tid >> 5;
  const int half = lane >> 4;
  const int mrow = lane & 15;
  for (int cc = 0; cc < 2; ++cc) {
    int cap = wave + cc * 8;            // 8 waves x 2 caps = 16 capsules
    float dot[8];
#pragma unroll
    for (int r = 0; r < 8; ++r) dot[r] = 0.f;
    for (int q = 0; q < 4; ++q) {       // 4 col-tiles per capsule (64 dims)
      int ct = cap * 4 + q;
      int col = ct * 16 + mrow;
      v8f acc = wmma_tile(hT, Wv, bias, ct, lane);
#pragma unroll
      for (int r = 0; r < 8; ++r)
        dot[r] += acc[r] * v[gT[r + half * 8] * COLS + col];
    }
#pragma unroll
    for (int r = 0; r < 8; ++r) {       // reduce over 16 cols (lanes within half)
      dot[r] += __shfl_xor(dot[r], 1, 32);
      dot[r] += __shfl_xor(dot[r], 2, 32);
      dot[r] += __shfl_xor(dot[r], 4, 32);
      dot[r] += __shfl_xor(dot[r], 8, 32);
    }
    if (mrow == 0) {
#pragma unroll
      for (int r = 0; r < 8; ++r) {
        int node = nodebase + r + half * 8;
        blog[node * TC + cap] += dot[r];   // unique owner: non-atomic RMW
      }
    }
  }
}

// ---------- squash + finalize ----------
__global__ void k_squash(const float* __restrict__ s, float* __restrict__ v) {
  int id = blockIdx.x * blockDim.x + threadIdx.x;  // NG*TC = 4096
  if (id >= NG * TC) return;
  const float* sr = s + id * CH;
  float sq = 0.f;
  for (int d = 0; d < CH; ++d) sq += sr[d] * sr[d];
  float f = sq / ((1.f + sq) * sqrtf(sq + 1e-16f));
  float* vr = v + id * CH;
  for (int d = 0; d < CH; ++d) vr[d] = sr[d] * f;
}

__global__ void k_final(const float* __restrict__ s, const float* __restrict__ xsum,
                        const float* __restrict__ cnt, float* __restrict__ out) {
  int id = blockIdx.x * blockDim.x + threadIdx.x;
  if (id >= NG * TC) return;
  int g = id >> 4;
  const float* sr = s + id * CH;
  const float* xs = xsum + g * CH;
  float ic = 1.f / fmaxf(cnt[g], 1.f);
  float sq = 0.f;
  for (int d = 0; d < CH; ++d) { float sv = sr[d] + xs[d] * ic; sq += sv * sv; }
  float f = sq / ((1.f + sq) * sqrtf(sq + 1e-16f));
  float acc = 0.f;
  for (int d = 0; d < CH; ++d) {
    float sv = sr[d] + xs[d] * ic;
    acc += 1.f / fabsf(sv * f);
  }
  out[id] = 1.f / acc;
}

// ---------- host ----------
extern "C" void kernel_launch(void* const* d_in, const int* in_sizes, int n_in,
                              void* d_out, int out_size, void* d_ws, size_t ws_size,
                              hipStream_t stream) {
  (void)in_sizes; (void)n_in; (void)out_size; (void)ws_size;
  const float* x     = (const float*)d_in[0];
  const int*   ei    = (const int*)d_in[1];
  const float* ew    = (const float*)d_in[2];
  const int*   batch = (const int*)d_in[3];
  const float* gam   = (const float*)d_in[4];
  const float* bet   = (const float*)d_in[5];
  const float* W     = (const float*)d_in[6];
  const float* bias  = (const float*)d_in[7];
  float* out = (float*)d_out;
  const int* row = ei;
  const int* col = ei + NE;

  char* base = (char*)d_ws;
  size_t off = 0;
  auto take = [&](size_t bytes) -> char* {
    char* p = base + off;
    off = (off + bytes + 255) & ~(size_t)255;
    return p;
  };
  float* partial = (float*)take(128 * 4);
  float* scale   = (float*)take(CH * 4);
  float* shiftv  = (float*)take(CH * 4);
  float* xn      = (float*)take((size_t)NN * CH * 4);
  float* deg     = (float*)take((size_t)NN * 4);
  float* ndeg    = (float*)take((size_t)NN * 4);
  float* dinv    = (float*)take((size_t)NN * 4);
  float* h       = (float*)take((size_t)NN * CH * 4);
  unsigned short* Wswz = (unsigned short*)take((size_t)CH * COLS * 2);
  float* blog    = (float*)take((size_t)NN * TC * 4);
  float* s       = (float*)take((size_t)NG * COLS * 4);
  float* v       = (float*)take((size_t)NG * COLS * 4);
  float* xsum    = (float*)take((size_t)NG * CH * 4);
  float* cnt     = (float*)take((size_t)NG * 4);
  const v16bf* Wv = (const v16bf*)Wswz;

  hipMemsetAsync(partial, 0, 128 * 4, stream);
  hipMemsetAsync(ndeg, 0, (size_t)NN * 4, stream);
  hipMemsetAsync(xsum, 0, (size_t)NG * CH * 4, stream);
  hipMemsetAsync(cnt, 0, (size_t)NG * 4, stream);

  k_bn_stats<<<256, 256, 0, stream>>>(x, partial);
  k_bn_final<<<1, 64, 0, stream>>>(partial, gam, bet, scale, shiftv);
  k_xn<<<(NN * CH) / 256, 256, 0, stream>>>(x, scale, shiftv, xn);
  k_deg_init<<<(NN + 255) / 256, 256, 0, stream>>>(deg);
  k_edge_deg<<<NE / 256, 256, 0, stream>>>(row, col, ew, deg, ndeg);
  k_dinv<<<(NN + 255) / 256, 256, 0, stream>>>(deg, dinv);
  k_hself<<<(NN * CH) / 256, 256, 0, stream>>>(xn, dinv, h);
  k_hscatter<<<(NE * CH) / 256, 256, 0, stream>>>(row, col, ew, dinv, xn, h);
  k_wconv<<<(TC * CH * CH) / 256, 256, 0, stream>>>(W, Wswz);
  k_binit<<<(NN * TC) / 256, 256, 0, stream>>>(ndeg, blog);
  k_xsum<<<(NN * CH) / 256, 256, 0, stream>>>(xn, batch, xsum, cnt);

  for (int it = 0; it < 2; ++it) {   // NUM_ROUTES - 1
    hipMemsetAsync(s, 0, (size_t)NG * COLS * 4, stream);
    k_route_s<<<NN / 16, 256, 0, stream>>>(h, Wv, bias, blog, batch, s);
    k_squash<<<(NG * TC) / 256, 256, 0, stream>>>(s, v);
    k_route_b<<<NN / 16, 256, 0, stream>>>(h, Wv, bias, v, batch, blog);
  }
  hipMemsetAsync(s, 0, (size_t)NG * COLS * 4, stream);
  k_route_s<<<NN / 16, 256, 0, stream>>>(h, Wv, bias, blog, batch, s);
  k_final<<<(NG * TC) / 256, 256, 0, stream>>>(s, xsum, cnt, out);
}
